// BipartiteGAT_29652454212572
// MI455X (gfx1250) — matrix operand order repaired and checked by reading, not compiled, not run on hardware
//
#include <hip/hip_runtime.h>
#include <math.h>

// Problem constants (from reference)
#define N_SRCC 50000
#define N_DSTC 25000
#define NE     200000
#define DD     128
#define HH     4
#define HC     512   // H*D
#define EDGED  32
#define KK     128   // GEMM K (hidden dim)
#define NEG_SLOPE 0.2f
#define LN_EPS 1e-5f

// padded row counts (multiple of 64) so GEMM tiles never read OOB
#define N_SRCP 50048
#define N_DSTP 25024

typedef __attribute__((ext_vector_type(16))) __bf16 v16bf;
typedef __attribute__((ext_vector_type(8)))  __bf16 v8bf;
typedef __attribute__((ext_vector_type(8)))  float  v8f;
typedef __attribute__((ext_vector_type(4)))  unsigned uint4v;
typedef __attribute__((ext_vector_type(4)))  int      v4i;

__device__ __forceinline__ unsigned short f2bfbits(float f) {
  unsigned u = __float_as_uint(f);
  u += 0x7FFFu + ((u >> 16) & 1u);          // round-to-nearest-even
  return (unsigned short)(u >> 16);
}
__device__ __forceinline__ unsigned fenc(float f) {  // monotone float->u32
  unsigned u = __float_as_uint(f);
  return (u & 0x80000000u) ? ~u : (u | 0x80000000u);
}
__device__ __forceinline__ float fdec(unsigned e) {
  unsigned u = (e & 0x80000000u) ? (e & 0x7FFFFFFFu) : ~e;
  return __uint_as_float(u);
}
__device__ __forceinline__ float lrelu(float x) {
  return fmaxf(x, 0.f) + NEG_SLOPE * fminf(x, 0.f);
}

// --- CDNA5 async global->LDS staging (16B per lane), with safe fallback ----
__device__ __forceinline__ void async_copy16(const unsigned short* g, unsigned short* l) {
#if __has_builtin(__builtin_amdgcn_global_load_async_to_lds_b128)
  __builtin_amdgcn_global_load_async_to_lds_b128(
      (__attribute__((address_space(1))) v4i*)g,
      (__attribute__((address_space(3))) v4i*)l, 0, 0);
#else
  *(uint4v*)l = *(const uint4v*)g;
#endif
}
__device__ __forceinline__ void wait_async0() {
#if __has_builtin(__builtin_amdgcn_s_wait_asynccnt)
  __builtin_amdgcn_s_wait_asynccnt(0);
#elif __has_builtin(__builtin_amdgcn_global_load_async_to_lds_b128)
  asm volatile("s_wait_asynccnt 0x0" ::: "memory");
#endif
}

// fragment loaders: contiguous k order (K = 16*(lane/16) + i) for both A and B
__device__ __forceinline__ v16bf load_afrag(const unsigned short* p /*16B-aligned, 16 halfs*/) {
  // A: element i <-> K = i (+16*(lane/16) folded into p by caller)
  v8bf lo = __builtin_bit_cast(v8bf, *(const uint4v*)p);
  v8bf hi = __builtin_bit_cast(v8bf, *(const uint4v*)(p + 8));
  v16bf a;
#pragma unroll
  for (int i = 0; i < 8; i++) { a[i] = lo[i]; a[i + 8] = hi[i]; }
  return a;
}
__device__ __forceinline__ v16bf load_bfrag(const unsigned short* p /*16 contiguous halfs*/) {
  // B: element i <-> K = i (+16*(lane/16) folded into p by caller)
  v8bf lo = __builtin_bit_cast(v8bf, *(const uint4v*)p);
  v8bf hi = __builtin_bit_cast(v8bf, *(const uint4v*)(p + 8));
  v16bf b;
#pragma unroll
  for (int i = 0; i < 8; i++) { b[i] = lo[i]; b[i + 8] = hi[i]; }
  return b;
}

// ---------------------------------------------------------------------------
// f32 -> bf16 bulk conversion (4 elems/thread)
// ---------------------------------------------------------------------------
__global__ __launch_bounds__(256)
void f32_to_bf16_kernel(const float* __restrict__ in, unsigned short* __restrict__ o, int n4) {
  int i = blockIdx.x * blockDim.x + threadIdx.x;
  if (i >= n4) return;
  const float4 v = ((const float4*)in)[i];
  unsigned q0 = f2bfbits(v.x) | ((unsigned)f2bfbits(v.y) << 16);
  unsigned q1 = f2bfbits(v.z) | ((unsigned)f2bfbits(v.w) << 16);
  ((uint2*)o)[i] = make_uint2(q0, q1);
}

// ---------------------------------------------------------------------------
// C[M][512] = A[M][128] @ W[512][128]^T + bias[512]  (bf16 in, f32 out)
// Block 256 thr = 8 waves; tile 64(M) x 64(N); each wave a 16x32 slab
// -> 2 WMMA per k-step, 8 per wave (K fully unrolled).
// A/W are pre-converted bf16 with padded rows (no OOB reads; stores guarded).
// Staging uses GLOBAL_LOAD_ASYNC_TO_LDS_B128 (ASYNCcnt) when available.
// ---------------------------------------------------------------------------
__global__ __launch_bounds__(256)
void gemm_xw_bias(const unsigned short* __restrict__ Ab, const unsigned short* __restrict__ Wb,
                  const float* __restrict__ bias, float* __restrict__ C, int M) {
  __shared__ unsigned short As[64][40];  // +8 pad; 80B row stride (16B-multiple)
  __shared__ unsigned short Bs[64][40];
  const int tid  = threadIdx.x;
  const int wave = tid >> 5, lane = tid & 31;
  const int wm = wave & 3, wn = wave >> 2;   // 4 x 2 wave grid
  const int m0 = blockIdx.x * 64;
  const int n0 = blockIdx.y * 64;
  const int hl = lane >> 4;
  const int r = tid >> 2;                    // staging row (0..63)
  const int c = (tid & 3) * 8;               // staging col in halfs {0,8,16,24}
  v8f acc0 = {}, acc1 = {};
#pragma unroll
  for (int kk = 0; kk < KK; kk += 32) {
    async_copy16(Ab + (size_t)(m0 + r) * KK + kk + c, &As[r][c]);
    async_copy16(Wb + (size_t)(n0 + r) * KK + kk + c, &Bs[r][c]);
    wait_async0();
    __syncthreads();
    const v16bf a  = load_afrag(&As[wm * 16 + (lane & 15)][16 * hl]);
    const v16bf b0 = load_bfrag(&Bs[wn * 32 + (lane & 15)][16 * hl]);
    const v16bf b1 = load_bfrag(&Bs[wn * 32 + 16 + (lane & 15)][16 * hl]);
    acc0 = __builtin_amdgcn_wmma_f32_16x16x32_bf16(false, a, false, b0, (short)0, acc0, false, false);
    acc1 = __builtin_amdgcn_wmma_f32_16x16x32_bf16(false, a, false, b1, (short)0, acc1, false, false);
    __syncthreads();
  }
  // D layout: m = vgpr + 8*(lane/16), n = lane%16
  const int nA = n0 + wn * 32 + (lane & 15);
  const int nB = nA + 16;
  const float bvA = bias[nA], bvB = bias[nB];
#pragma unroll
  for (int q = 0; q < 8; q++) {
    int m = m0 + wm * 16 + q + 8 * hl;
    if (m < M) {
      C[(size_t)m * HC + nA] = acc0[q] + bvA;
      C[(size_t)m * HC + nB] = acc1[q] + bvB;
    }
  }
}

// ---------------------------------------------------------------------------
// Fused edge kernel: xe = edge_rep @ We^T via WMMA (K=32 -> 1 wmma / n-tile),
// then alpha[e,h] = att[h] . leaky_relu(xl[src]+xr[dst]+xe), atomicMax amax.
// Block 256 thr = 8 waves, 16 edges. Fragments loaded straight from bf16
// global buffers (no staging, no conversion).
// ---------------------------------------------------------------------------
__global__ __launch_bounds__(256)
void edge_alpha_kernel(const unsigned short* __restrict__ Eb,
                       const unsigned short* __restrict__ WeB,
                       const float* __restrict__ att,
                       const float* __restrict__ xl, const float* __restrict__ xr,
                       const int* __restrict__ srcI, const int* __restrict__ dstI,
                       float* __restrict__ alphaB, unsigned* __restrict__ amaxE) {
  __shared__ int sSrc[16], sDst[16];
  __shared__ float xeS[16][516];  // xe tile f32 (+4 pad)
  __shared__ float attS[HC];
  const int tid = threadIdx.x;
  const int e0 = blockIdx.x * 16;
  if (tid < 16) { sSrc[tid] = srcI[e0 + tid]; sDst[tid] = dstI[e0 + tid]; }
  attS[tid] = att[tid];
  attS[tid + 256] = att[tid + 256];

  const int wave = tid >> 5, lane = tid & 31, hl = lane >> 4;
  // A frag: 16 edges x 32 edge-dims (row = e0 + lane%16)
  const v16bf a = load_afrag(Eb + (size_t)(e0 + (lane & 15)) * EDGED + 16 * hl);
#pragma unroll
  for (int t = 0; t < 4; t++) {
    const int n = wave * 64 + t * 16 + (lane & 15);
    const v16bf b = load_bfrag(WeB + (size_t)n * EDGED + 16 * hl);
    v8f cacc = {};
    cacc = __builtin_amdgcn_wmma_f32_16x16x32_bf16(false, a, false, b, (short)0, cacc, false, false);
#pragma unroll
    for (int q = 0; q < 8; q++) xeS[q + 8 * hl][n] = cacc[q];
  }
  __syncthreads();

  // 64 (edge,head) pairs; each wave does 8; each lane covers 4 channels.
#pragma unroll
  for (int j = 0; j < 8; j++) {
    const int p = wave * 8 + j;
    const int el = p >> 2, h = p & 3;
    const int s = sSrc[el], dn = sDst[el];
    const int col = h * DD + lane * 4;
    const float4 xlv = *(const float4*)(xl + (size_t)s * HC + col);
    const float4 xrv = *(const float4*)(xr + (size_t)dn * HC + col);
    const float4 xev = *(const float4*)&xeS[el][col];
    const float4 av = *(const float4*)&attS[col];
    float sum = av.x * lrelu(xlv.x + xrv.x + xev.x)
              + av.y * lrelu(xlv.y + xrv.y + xev.y)
              + av.z * lrelu(xlv.z + xrv.z + xev.z)
              + av.w * lrelu(xlv.w + xrv.w + xev.w);
#pragma unroll
    for (int m = 16; m >= 1; m >>= 1) sum += __shfl_xor(sum, m, 32);
    if (lane == 0) {
      alphaB[(size_t)(e0 + el) * HH + h] = sum;
      atomicMax(&amaxE[(size_t)dn * HH + h], fenc(sum));
    }
  }
}

// exp(alpha - amax[dst]); accumulate denom
__global__ __launch_bounds__(256)
void edge_softmax_kernel(float* __restrict__ alphaB, const int* __restrict__ dstI,
                         const unsigned* __restrict__ amaxE, float* __restrict__ denom) {
  int i = blockIdx.x * blockDim.x + threadIdx.x;
  if (i >= NE * HH) return;
  int e = i >> 2, h = i & 3;
  int dn = dstI[e];
  float ex = __expf(alphaB[i] - fdec(amaxE[dn * HH + h]));
  alphaB[i] = ex;
  unsafeAtomicAdd(&denom[dn * HH + h], ex);
}

// acc[dst,h,:] += (ex/denom) * xl[src,h,:]
__global__ __launch_bounds__(128)
void edge_scatter_kernel(const float* __restrict__ xl, const int* __restrict__ srcI,
                         const int* __restrict__ dstI, const float* __restrict__ alphaB,
                         const float* __restrict__ denom, float* __restrict__ acc) {
  const int e = blockIdx.x, t = threadIdx.x;
  const int s = srcI[e], dn = dstI[e];
  const int idx = t * 4, h = idx >> 7;
  const float w = alphaB[(size_t)e * HH + h] / (denom[dn * HH + h] + 1e-16f);
  const float4 xv = *(const float4*)(xl + (size_t)s * HC + idx);
  float* ap = acc + (size_t)dn * HC + idx;
  unsafeAtomicAdd(ap + 0, w * xv.x);
  unsafeAtomicAdd(ap + 1, w * xv.y);
  unsafeAtomicAdd(ap + 2, w * xv.z);
  unsafeAtomicAdd(ap + 3, w * xv.w);
}

// head-mean + bias + residual + LayerNorm (wave32 shuffle reductions)
__global__ __launch_bounds__(128)
void finalize_kernel(const float* __restrict__ acc, const float* __restrict__ bias,
                     const float* __restrict__ res, const float* __restrict__ g,
                     const float* __restrict__ b, float* __restrict__ out) {
  const int n = blockIdx.x, d = threadIdx.x;
  const float* ap = acc + (size_t)n * HC;
  float v = 0.25f * (ap[d] + ap[DD + d] + ap[2 * DD + d] + ap[3 * DD + d])
          + bias[d] + res[(size_t)n * DD + d];
  __shared__ float red[4], red2[4];
  const int wave = d >> 5, lane = d & 31;
  float s = v;
#pragma unroll
  for (int m = 16; m >= 1; m >>= 1) s += __shfl_xor(s, m, 32);
  if (lane == 0) red[wave] = s;
  __syncthreads();
  const float mean = (red[0] + red[1] + red[2] + red[3]) * (1.0f / DD);
  const float cdev = v - mean;
  float q = cdev * cdev;
#pragma unroll
  for (int m = 16; m >= 1; m >>= 1) q += __shfl_xor(q, m, 32);
  if (lane == 0) red2[wave] = q;
  __syncthreads();
  const float var = (red2[0] + red2[1] + red2[2] + red2[3]) * (1.0f / DD);
  out[(size_t)n * DD + d] = cdev * rsqrtf(var + LN_EPS) * g[d] + b[d];
}

// ---------------------------------------------------------------------------
extern "C" void kernel_launch(void* const* d_in, const int* in_sizes, int n_in,
                              void* d_out, int out_size, void* d_ws, size_t ws_size,
                              hipStream_t stream) {
  const float* send = (const float*)d_in[0];
  const float* rec  = (const float*)d_in[1];
  const float* erep = (const float*)d_in[2];
  const int*   eidx = (const int*)d_in[3];
  const int* srcI = eidx;
  const int* dstI = eidx + NE;

  // Param leaf order detection: insertion (Wl first, 65536) vs sorted (We first, 16384).
  // logical order: Wl, bl, Wr, br, We, att, bias, ln_g, ln_b
  static const int mapIns[9]  = {0, 1, 2, 3, 4, 5, 6, 7, 8};
  static const int mapSort[9] = {1, 5, 2, 6, 0, 3, 4, 8, 7};
  const int* map = (n_in > 4 && in_sizes[4] == HC * DD) ? mapIns : mapSort;

  // Workspace carve-up (~261 MB, all chunks 128B-aligned)
  char* w = (char*)d_ws;
  float* xl       = (float*)w;          w += (size_t)N_SRCC * HC * 4;
  float* xr       = (float*)w;          w += (size_t)N_DSTC * HC * 4;
  float* alphaB   = (float*)w;          w += (size_t)NE * HH * 4;
  unsigned* amaxE = (unsigned*)w;       w += (size_t)N_DSTC * HH * 4;
  float* denom    = (float*)w;          w += (size_t)N_DSTC * HH * 4;
  float* acc      = (float*)w;          w += (size_t)N_DSTC * HC * 4;
  float* h1       = (float*)w;          w += (size_t)N_DSTC * DD * 4;
  unsigned short* sendB = (unsigned short*)w; w += (size_t)N_SRCP * KK * 2;
  unsigned short* recB  = (unsigned short*)w; w += (size_t)N_DSTP * KK * 2;
  unsigned short* h1B   = (unsigned short*)w; w += (size_t)N_DSTP * KK * 2;
  unsigned short* EbB   = (unsigned short*)w; w += (size_t)NE * EDGED * 2;
  unsigned short* WlB   = (unsigned short*)w; w += (size_t)HC * KK * 2;
  unsigned short* WrB   = (unsigned short*)w; w += (size_t)HC * KK * 2;
  unsigned short* WeB   = (unsigned short*)w; w += (size_t)HC * EDGED * 2;
  (void)ws_size; (void)out_size;

  // one-time bf16 conversions of activations / edge features
  f32_to_bf16_kernel<<<(N_SRCC * KK / 4 + 255) / 256, 256, 0, stream>>>(send, sendB, N_SRCC * KK / 4);
  f32_to_bf16_kernel<<<(N_DSTC * KK / 4 + 255) / 256, 256, 0, stream>>>(rec, recB, N_DSTC * KK / 4);
  f32_to_bf16_kernel<<<(NE * EDGED / 4 + 255) / 256, 256, 0, stream>>>(erep, EbB, NE * EDGED / 4);

  const unsigned short* xdstB = recB;  // bf16 x_dst for layer 1
  const float* resp = rec;             // residual for layer 1
  for (int L = 0; L < 2; L++) {
    const float* P[9];
    for (int j = 0; j < 9; j++) P[j] = (const float*)d_in[4 + L * 9 + map[j]];
    const float *Wl = P[0], *bl = P[1], *Wr = P[2], *br = P[3], *We = P[4],
                *att = P[5], *bias = P[6], *lng = P[7], *lnb = P[8];
    float* outp = (L == 0) ? h1 : (float*)d_out;

    // per-layer weight conversions (tiny)
    f32_to_bf16_kernel<<<(HC * KK / 4 + 255) / 256, 256, 0, stream>>>(Wl, WlB, HC * KK / 4);
    f32_to_bf16_kernel<<<(HC * KK / 4 + 255) / 256, 256, 0, stream>>>(Wr, WrB, HC * KK / 4);
    f32_to_bf16_kernel<<<(HC * EDGED / 4 + 255) / 256, 256, 0, stream>>>(We, WeB, HC * EDGED / 4);

    // x_src stays send_rep in the reference for BOTH layers.
    gemm_xw_bias<<<dim3(N_SRCP / 64, HC / 64), 256, 0, stream>>>(sendB, WlB, bl, xl, N_SRCC);
    gemm_xw_bias<<<dim3(N_DSTP / 64, HC / 64), 256, 0, stream>>>(xdstB, WrB, br, xr, N_DSTC);

    (void)hipMemsetAsync(amaxE, 0, (size_t)N_DSTC * HH * 4, stream);
    (void)hipMemsetAsync(denom, 0, (size_t)N_DSTC * HH * 4, stream);
    (void)hipMemsetAsync(acc, 0, (size_t)N_DSTC * HC * 4, stream);

    edge_alpha_kernel<<<NE / 16, 256, 0, stream>>>(EbB, WeB, att, xl, xr,
                                                   srcI, dstI, alphaB, amaxE);
    edge_softmax_kernel<<<(NE * HH + 255) / 256, 256, 0, stream>>>(alphaB, dstI, amaxE, denom);
    edge_scatter_kernel<<<NE, 128, 0, stream>>>(xl, srcI, dstI, alphaB, denom, acc);
    finalize_kernel<<<N_DSTC, 128, 0, stream>>>(acc, bias, resp, lng, lnb, outp);

    if (L == 0) {  // prepare bf16 x_dst for layer 2
      f32_to_bf16_kernel<<<(N_DSTC * KK / 4 + 255) / 256, 256, 0, stream>>>(h1, h1B, N_DSTC * KK / 4);
      xdstB = h1B;
      resp = h1;
    }
  }
}
